// FrequencyBandModulation_62818191671614
// MI455X (gfx1250) — compile-verified, matrix operand order checked by reading, not run on hardware
//
#include <hip/hip_runtime.h>
#include <cstdint>

// ---------------- CDNA5 WMMA / TDM types ----------------
typedef __attribute__((ext_vector_type(16))) __bf16 v16bf;
typedef __attribute__((ext_vector_type(8)))  float  v8f;
typedef __attribute__((ext_vector_type(4)))  unsigned int u32x4;
typedef __attribute__((ext_vector_type(8)))  int i32x8;
typedef __attribute__((ext_vector_type(4)))  int i32x4;

#define WMMA_BF16(a, b, c) \
  __builtin_amdgcn_wmma_f32_16x16x32_bf16(false, (a), false, (b), (short)0, (c), false, false)

union FragU { v16bf v; uint32_t u[8]; u32x4 q[2]; };

// Load a 16x32 bf16 fragment (A-style layout, ISA 7.12.2) from a row-major
// bf16 matrix with leading dimension ld (elements). lanes 0-15: rows M,
// K 0-7/16-23; lanes 16-31: rows M, K 8-15/24-31. Two 16B loads per lane;
// offsets are 16B aligned (ld=256, k0%32==0, hi*8). Works for global or LDS
// source (compiler lowers to global_load_b128 / ds_load_b128 respectively).
__device__ __forceinline__ v16bf load_frag(const __bf16* base, int row0, int ld, int k0) {
  const int lane = threadIdx.x & 31;
  const int m = lane & 15, hi = lane >> 4;
  const uint16_t* p = (const uint16_t*)base + (size_t)(row0 + m) * ld + k0 + 8 * hi;
  FragU f;
  f.q[0] = *(const u32x4*)(p);
  f.q[1] = *(const u32x4*)(p + 16);
  return f.v;
}

// Negate a bf16 fragment by flipping packed sign bits (cheaper than reloading).
__device__ __forceinline__ v16bf neg_frag(v16bf a) {
  FragU f; f.v = a;
#pragma unroll
  for (int i = 0; i < 8; ++i) f.u[i] ^= 0x80008000u;
  return f.v;
}

// C/D layout (ISA 7.12.2): VGPR r, lane l -> row = r + 8*(l>>4), col = l&15.
__device__ __forceinline__ void store_tile_bf16(__bf16* dst, int ld, int row0, int col0, v8f acc) {
  const int lane = threadIdx.x & 31;
  const int hi = lane >> 4, n = lane & 15;
#pragma unroll
  for (int r = 0; r < 8; ++r)
    dst[(size_t)(row0 + r + 8 * hi) * ld + col0 + n] = (__bf16)acc[r];
}

// ---------------- Tensor Data Mover: 16x256 bf16 tile -> LDS ----------------
// D# per CDNA5 ISA: 2D tensor (groups 2/3 zero),
// data_size=2B, tile_dim0=256 (row, contiguous 512B), tile_dim1=16 rows,
// tensor_dim0=tensor_dim1=256, tensor_dim0_stride=256, tensor_dim1_stride=64K.
__device__ __forceinline__ void tdm_load_tile_16x256(const __bf16* gsrc, __bf16* lds_dst) {
  const unsigned long long ga = (unsigned long long)(uintptr_t)gsrc;
  u32x4 g0;
  g0[0] = 1u;                                   // count=1, user descriptor
  g0[1] = (unsigned int)(uintptr_t)lds_dst;     // lds_addr = addr[31:0]
  g0[2] = (unsigned int)ga;                     // global_addr[31:0]
  g0[3] = (unsigned int)((ga >> 32) & 0x01FFFFFFu) | 0x80000000u;  // addr[56:32], type=2
  i32x8 g1;
  g1[0] = (int)(1u << 16);       // workgroup_mask=0, data_size=1 (2 bytes)
  g1[1] = (int)(256u << 16);     // tensor_dim0 = 256  (bits 79:48)
  g1[2] = (int)(256u << 16);     // tensor_dim1 = 256  (bits 111:80)
  g1[3] = (int)(256u << 16);     // tile_dim0  = 256  (bits 127:112)
  g1[4] = 16;                    // tile_dim1  = 16, tile_dim2 = 0
  g1[5] = 256;                   // tensor_dim0_stride = 256 (bits 207:160)
  g1[6] = 0;
  g1[7] = 1;                     // tensor_dim1_stride = 65536 (bits 255:208)
  i32x4 z4 = {0, 0, 0, 0};
#if __clang_major__ >= 23
  i32x8 z8 = {0, 0, 0, 0, 0, 0, 0, 0};
  __builtin_amdgcn_tensor_load_to_lds(g0, g1, z4, z4, z8, 0);
#else
  __builtin_amdgcn_tensor_load_to_lds(g0, g1, z4, z4, 0);
#endif
}

// ---------------- Kernel 1: DFT tables ----------------
// D[j,k] = exp(-2*pi*i*j*k/256)/16  (ortho 1D scale). D is symmetric.
__global__ void init_dft(__bf16* cre, __bf16* cim) {
  const int j = blockIdx.x, k = threadIdx.x;
  const int p = (j * k) & 255;
  const float th = (float)p * (6.283185307179586f / 256.0f);
  cre[j * 256 + k] = (__bf16)(cosf(th) * 0.0625f);
  cim[j * 256 + k] = (__bf16)(-sinf(th) * 0.0625f);
}

// ---------------- Kernel 2: global avg+max pool per (b,c) ----------------
__global__ void pool_kernel(const float* __restrict__ x, float* avg, float* mx) {
  const int img = blockIdx.x;                 // 0..511
  const int tid = threadIdx.x;                // 256 threads
  const float4* p = (const float4*)(x + (size_t)img * 65536);
  float s = 0.f, m = -3.402823466e38f;
  for (int i = tid; i < 16384; i += 256) {
    float4 v = p[i];
    s += v.x + v.y + v.z + v.w;
    m = fmaxf(m, fmaxf(fmaxf(v.x, v.y), fmaxf(v.z, v.w)));
  }
  __shared__ float ss[256], sm[256];
  ss[tid] = s; sm[tid] = m;
  __syncthreads();
  for (int o = 128; o > 0; o >>= 1) {
    if (tid < o) { ss[tid] += ss[tid + o]; sm[tid] = fmaxf(sm[tid], sm[tid + o]); }
    __syncthreads();
  }
  if (tid == 0) { avg[img] = ss[0] * (1.f / 65536.f); mx[img] = sm[0]; }
}

// ---------------- Kernel 3: channel-attention MLPs ----------------
__global__ void ca_kernel(const float* avg, const float* mxp,
                          const float* camw, const float* camb, const float* cg, const float* cbe,
                          const float* cw1, const float* cb1, const float* cw2, const float* cb2,
                          const float* bamw, const float* bamb, const float* bg, const float* bbe,
                          const float* bw1, const float* bb1, const float* bw2, const float* bb2,
                          float* cum, float* filt) {
  const int b = blockIdx.x, c = threadIdx.x;   // 8 blocks x 64 threads
  __shared__ float am[64], hh[64], oo[4];
  const float a = avg[b * 64 + c];
  const float m = mxp[b * 64 + c];
  const float bninv = rsqrtf(1.f + 1e-5f);
  for (int br = 0; br < 2; ++br) {
    const float* amw = br ? bamw : camw;  const float* amb = br ? bamb : camb;
    const float* g   = br ? bg   : cg;    const float* be  = br ? bbe  : cbe;
    const float* w1  = br ? bw1  : cw1;   const float* b1  = br ? bb1  : cb1;
    const float* w2  = br ? bw2  : cw2;   const float* b2  = br ? bb2  : cb2;
    float t = amw[c * 2] * a + amw[c * 2 + 1] * m + amb[c];
    t = g[c] * t * bninv + be[c];
    t = t > 0.f ? t : 0.2f * t;
    am[c] = t;
    __syncthreads();
    float acc = b1[c];
    for (int k = 0; k < 64; ++k) acc += w1[c * 64 + k] * am[k];
    hh[c] = acc > 0.f ? acc : 0.2f * acc;
    __syncthreads();
    if (c < 4) {
      float o = b2[c];
      for (int k = 0; k < 64; ++k) o += w2[c * 64 + k] * hh[k];
      oo[c] = o;
    }
    __syncthreads();
    if (c == 0) {
      float mo = fmaxf(fmaxf(oo[0], oo[1]), fmaxf(oo[2], oo[3]));
      float e[4], se = 0.f;
      for (int n = 0; n < 4; ++n) { e[n] = expf(oo[n] - mo); se += e[n]; }
      if (br == 0) {
        float cs = 0.f;
        for (int n = 0; n < 4; ++n) { cs += e[n] / se * 0.5f; cum[b * 4 + n] = cs; }
      } else {
        for (int n = 0; n < 4; ++n) filt[b * 4 + n] = e[n] / se * 4.0f;
      }
    }
    __syncthreads();
  }
}

// ---------------- Kernel 4: f32 -> bf16 transposed copy of x ----------------
__global__ void transpose_x(const float* __restrict__ x, __bf16* xt, int img0) {
  const int i = blockIdx.z;
  const int h0 = blockIdx.y * 16, w0 = blockIdx.x * 16;
  __shared__ float t[16][17];
  const float* src = x + (size_t)(img0 + i) * 65536;
  t[threadIdx.y][threadIdx.x] = src[(size_t)(h0 + threadIdx.y) * 256 + w0 + threadIdx.x];
  __syncthreads();
  __bf16* dst = xt + (size_t)i * 65536;
  dst[(size_t)(w0 + threadIdx.y) * 256 + h0 + threadIdx.x] = (__bf16)t[threadIdx.x][threadIdx.y];
}

// Tile indexing: 256 threads = 8 waves; 32 blocks per image.
// All 8 waves of a block share tm (so A operands are block-uniform and can be
// TDM-staged into LDS once); tn = (g&1)*8 + wave.
#define TILE_SETUP()                                   \
  const int i  = blockIdx.x >> 5;                      \
  const int g  = blockIdx.x & 31;                      \
  const int wv = threadIdx.x >> 5;                     \
  const int tm = g >> 1;                               \
  const int tn = (g & 1) * 8 + wv;                     \
  (void)tm; (void)tn;

// Stage the two block-uniform A slabs (16x256 bf16 each) into LDS via TDM.
#define STAGE_A_TO_LDS(a0src, a1src)                                   \
  __shared__ __bf16 lA0[16 * 256], lA1[16 * 256];                      \
  if (wv == 0) {                                                       \
    tdm_load_tile_16x256((a0src) + (size_t)tm * 16 * 256, lA0);        \
    tdm_load_tile_16x256((a1src) + (size_t)tm * 16 * 256, lA1);        \
    __builtin_amdgcn_s_wait_tensorcnt(0);                              \
  }                                                                    \
  __syncthreads();

// ---------------- Stage 1: Y = D @ x  (S=Re, T=Im) ----------------
__global__ void stage1(const __bf16* cre, const __bf16* cim, const __bf16* xt,
                       __bf16* sre, __bf16* sim) {
  TILE_SETUP();
  STAGE_A_TO_LDS(cre, cim);                       // A = Cre,Cim rows tm*16..+15
  const __bf16* xb = xt + (size_t)i * 65536;
  v8f aS = {}, aT = {};
#pragma unroll
  for (int k0 = 0; k0 < 256; k0 += 32) {
    v16bf fre = load_frag(lA0, 0, 256, k0);       // ds_load_b128 x2
    v16bf fim = load_frag(lA1, 0, 256, k0);
    v16bf fx  = load_frag(xb, tn * 16, 256, k0);  // global_load_b128 x2
    aS = WMMA_BF16(fre, fx, aS);
    aT = WMMA_BF16(fim, fx, aT);
  }
  store_tile_bf16(sre + (size_t)i * 65536, 256, tm * 16, tn * 16, aS);
  store_tile_bf16(sim + (size_t)i * 65536, 256, tm * 16, tn * 16, aT);
}

// ---------------- Stage 2: Z = Y @ D, apply band mask, store transposed ----------------
__global__ void stage2(const __bf16* cre, const __bf16* cim,
                       const __bf16* sre, const __bf16* sim,
                       const float* __restrict__ cum, const float* __restrict__ filt,
                       __bf16* ut, __bf16* vt, int img0) {
  TILE_SETUP();
  STAGE_A_TO_LDS(sre + (size_t)i * 65536, sim + (size_t)i * 65536);  // A = S,T rows
  v8f aU = {}, aV = {};
#pragma unroll
  for (int k0 = 0; k0 < 256; k0 += 32) {
    v16bf fs  = load_frag(lA0, 0, 256, k0);
    v16bf ft  = load_frag(lA1, 0, 256, k0);
    v16bf fre = load_frag(cre, tn * 16, 256, k0);  // D symmetric -> row-contig B
    v16bf fim = load_frag(cim, tn * 16, 256, k0);
    v16bf fmi = neg_frag(fim);
    aU = WMMA_BF16(fs, fre, aU);  aU = WMMA_BF16(ft, fmi, aU);  // U = S*Cre - T*Cim
    aV = WMMA_BF16(fs, fim, aV);  aV = WMMA_BF16(ft, fre, aV);  // V = S*Cim + T*Cre
  }
  const int b = (img0 + i) >> 6;  // C = 64
  const float c0 = cum[b * 4 + 0], c1 = cum[b * 4 + 1], c2 = cum[b * 4 + 2], c3 = cum[b * 4 + 3];
  const float f0 = filt[b * 4 + 0], f1 = filt[b * 4 + 1], f2 = filt[b * 4 + 2], f3 = filt[b * 4 + 3];
  const int lane = threadIdx.x & 31, hi = lane >> 4, n = lane & 15;
  __bf16* U = ut + (size_t)i * 65536;
  __bf16* V = vt + (size_t)i * 65536;
#pragma unroll
  for (int r = 0; r < 8; ++r) {
    const int row = tm * 16 + r + 8 * hi;     // h'
    const int col = tn * 16 + n;              // w'
    const float fh = (float)min(row, 256 - row) * (1.f / 256.f);
    const float fw = (float)min(col, 256 - col) * (1.f / 256.f);
    const float f = sqrtf(fh * fh + fw * fw);
    float comb = 0.f;                          // bands partition (lower_n, cum_n]
    if (f <= c3) comb = f3;
    if (f <= c2) comb = f2;
    if (f <= c1) comb = f1;
    if (f <= c0) comb = f0;
    U[(size_t)col * 256 + row] = (__bf16)(aU[r] * comb);  // transposed store
    V[(size_t)col * 256 + row] = (__bf16)(aV[r] * comb);
  }
}

// ---------------- Stage 3: A2 = conj(D) @ Z ----------------
__global__ void stage3(const __bf16* cre, const __bf16* cim,
                       const __bf16* ut, const __bf16* vt,
                       __bf16* pre, __bf16* pim) {
  TILE_SETUP();
  STAGE_A_TO_LDS(cre, cim);                       // A = Cre,Cim rows tm*16..+15
  const __bf16* U = ut + (size_t)i * 65536;       // stored [w'][h'] -> B contiguous
  const __bf16* V = vt + (size_t)i * 65536;
  v8f aP = {}, aQ = {};
#pragma unroll
  for (int k0 = 0; k0 < 256; k0 += 32) {
    v16bf fre = load_frag(lA0, 0, 256, k0);
    v16bf fim = load_frag(lA1, 0, 256, k0);
    v16bf fu  = load_frag(U, tn * 16, 256, k0);
    v16bf fv  = load_frag(V, tn * 16, 256, k0);
    v16bf fmi = neg_frag(fim);
    aP = WMMA_BF16(fre, fu, aP);  aP = WMMA_BF16(fim, fv, aP);  // P = Cre*U + Cim*V
    aQ = WMMA_BF16(fre, fv, aQ);  aQ = WMMA_BF16(fmi, fu, aQ);  // Q = Cre*V - Cim*U
  }
  store_tile_bf16(pre + (size_t)i * 65536, 256, tm * 16, tn * 16, aP);
  store_tile_bf16(pim + (size_t)i * 65536, 256, tm * 16, tn * 16, aQ);
}

// ---------------- Stage 4: out = Re(A2 @ conj(D)) ----------------
__global__ void stage4(const __bf16* cre, const __bf16* cim,
                       const __bf16* pre, const __bf16* pim,
                       float* __restrict__ out, int img0) {
  TILE_SETUP();
  STAGE_A_TO_LDS(pre + (size_t)i * 65536, pim + (size_t)i * 65536);  // A = P,Q rows
  v8f acc = {};
#pragma unroll
  for (int k0 = 0; k0 < 256; k0 += 32) {
    v16bf fp  = load_frag(lA0, 0, 256, k0);
    v16bf fq  = load_frag(lA1, 0, 256, k0);
    v16bf fre = load_frag(cre, tn * 16, 256, k0);
    v16bf fim = load_frag(cim, tn * 16, 256, k0);
    acc = WMMA_BF16(fp, fre, acc);   // out = P*Cre + Q*Cim  ( = P*Ere - Q*Eim )
    acc = WMMA_BF16(fq, fim, acc);
  }
  float* ob = out + (size_t)(img0 + i) * 65536;
  const int lane = threadIdx.x & 31, hi = lane >> 4, n = lane & 15;
#pragma unroll
  for (int r = 0; r < 8; ++r)
    ob[(size_t)(tm * 16 + r + 8 * hi) * 256 + tn * 16 + n] = acc[r];
}

// ---------------- Host launch ----------------
extern "C" void kernel_launch(void* const* d_in, const int* in_sizes, int n_in,
                              void* d_out, int out_size, void* d_ws, size_t ws_size,
                              hipStream_t stream) {
  const float* x = (const float*)d_in[0];
  const float* camw = (const float*)d_in[1];
  const float* camb = (const float*)d_in[2];
  const float* cg   = (const float*)d_in[3];
  const float* cbe  = (const float*)d_in[4];
  const float* cw1  = (const float*)d_in[5];
  const float* cb1  = (const float*)d_in[6];
  const float* cw2  = (const float*)d_in[7];
  const float* cb2  = (const float*)d_in[8];
  const float* bamw = (const float*)d_in[9];
  const float* bamb = (const float*)d_in[10];
  const float* bg   = (const float*)d_in[11];
  const float* bbe  = (const float*)d_in[12];
  const float* bw1  = (const float*)d_in[13];
  const float* bb1  = (const float*)d_in[14];
  const float* bw2  = (const float*)d_in[15];
  const float* bb2  = (const float*)d_in[16];
  float* out = (float*)d_out;

  char* ws = (char*)d_ws;
  __bf16* cre  = (__bf16*)(ws + 0);        // 128 KB
  __bf16* cim  = (__bf16*)(ws + 131072);   // 128 KB
  float*  avg  = (float*)(ws + 262144);    // 2 KB
  float*  mxp  = (float*)(ws + 264192);    // 2 KB
  float*  cum  = (float*)(ws + 266240);    // 128 B
  float*  filt = (float*)(ws + 266368);    // 128 B
  const size_t base = 270336;

  // Chunk size (images per pass), sized from ws: 5 bf16 image buffers per chunk.
  int NI = 32;
  while (NI > 1 && base + 5ull * (size_t)NI * 131072ull > ws_size) NI >>= 1;
  const size_t ib = (size_t)NI * 65536;    // elements per image buffer
  __bf16* xt  = (__bf16*)(ws + base);
  __bf16* sre = xt  + ib;
  __bf16* sim = sre + ib;
  __bf16* ut  = sim + ib;
  __bf16* vt  = ut  + ib;

  init_dft<<<256, 256, 0, stream>>>(cre, cim);
  pool_kernel<<<512, 256, 0, stream>>>(x, avg, mxp);
  ca_kernel<<<8, 64, 0, stream>>>(avg, mxp,
                                  camw, camb, cg, cbe, cw1, cb1, cw2, cb2,
                                  bamw, bamb, bg, bbe, bw1, bb1, bw2, bb2,
                                  cum, filt);

  for (int img0 = 0; img0 < 512; img0 += NI) {
    transpose_x<<<dim3(16, 16, NI), dim3(16, 16), 0, stream>>>(x, xt, img0);
    stage1<<<NI * 32, 256, 0, stream>>>(cre, cim, xt, sre, sim);
    stage2<<<NI * 32, 256, 0, stream>>>(cre, cim, sre, sim, cum, filt, ut, vt, img0);
    stage3<<<NI * 32, 256, 0, stream>>>(cre, cim, ut, vt, sre, sim);  // P,Q overwrite S,T
    stage4<<<NI * 32, 256, 0, stream>>>(cre, cim, sre, sim, out, img0);
  }
  (void)in_sizes; (void)n_in; (void)out_size;
}
